// BERT_12790412607911
// MI455X (gfx1250) — matrix operand order repaired and checked by reading, not compiled, not run on hardware
//
#include <hip/hip_runtime.h>

// ---------------------------------------------------------------------------
// Types for CDNA5 WMMA (wave32): v_wmma_f32_16x16x32_bf16
// ---------------------------------------------------------------------------
typedef __bf16 bf16_t;
typedef bf16_t v16bf __attribute__((ext_vector_type(16)));
typedef bf16_t v8bf  __attribute__((ext_vector_type(8)));
typedef bf16_t v4bf  __attribute__((ext_vector_type(4)));
typedef float  v8f   __attribute__((ext_vector_type(8)));
typedef float  v4f   __attribute__((ext_vector_type(4)));

#define BM 128
#define BK 32
#define LDSS 40   // 32 + 8 pad halves (80B rows: 16B-aligned frag reads)

#define FLAG_RELU 1

// ---------------------------------------------------------------------------
// Batched GEMM: C = act( scale * A@B + bias + Res ), bf16 WMMA, f32 accum.
// All dims must be exact tile multiples (M%128==0, N%TBN==0, K%32==0) --
// every launch below satisfies this, so there are NO bounds checks.
// TRANSB: B[k,n] = Bsrc[n*ldB + k]  (used for Q@K^T).
// Double-buffered LDS, one barrier per K-step.
// ---------------------------------------------------------------------------
template<int TBN, bool TRANSB>
__global__ __launch_bounds__(256)
void gemm_wmma_bf16(const float* __restrict__ A, const float* __restrict__ Bm,
                    float* __restrict__ C, const float* __restrict__ bias,
                    const float* __restrict__ Res,
                    int M, int N, int Kd, int ldA, int ldB, int ldC,
                    int batchInner,
                    long sAo, long sAi, long sBo, long sBi, long sCo, long sCi,
                    float scale, int flags)
{
    constexpr int NI = TBN / 32;           // wmma tiles per wave along N
    __shared__ __attribute__((aligned(16))) bf16_t As[2][BM * LDSS];
    __shared__ __attribute__((aligned(16))) bf16_t Bs[2][TBN * LDSS];

    const int tid  = threadIdx.x;
    const int lane = tid & 31;
    const int wave = tid >> 5;     // 0..7
    const int wm   = wave >> 1;    // 0..3 : 32 rows each
    const int wn   = wave & 1;     // 0..1 : TBN/2 cols each
    const int lh   = lane & 15;
    const int hb   = lane >> 4;    // lane-half

    const int n0 = blockIdx.x * TBN;
    const int m0 = blockIdx.y * BM;
    const int z  = blockIdx.z;
    const int zo = z / batchInner, zi = z % batchInner;
    const float* Ab = A  + zo * sAo + zi * sAi;
    const float* Bb = Bm + zo * sBo + zi * sBi;
    float*       Cb = C  + zo * sCo + zi * sCi;

    // ---- staging helpers (vectorized, unguarded) ----
    auto loadA = [&](int k0, v4f ar[4]) {
#pragma unroll
        for (int i = 0; i < 4; ++i) {
            int e4 = tid + i * 256;
            int r = e4 >> 3, c4 = e4 & 7;            // 8 float4 per 32-col row
            ar[i] = *(const v4f*)(Ab + (long)(m0 + r) * ldA + k0 + c4 * 4);
        }
    };
    auto storeA = [&](int p, const v4f ar[4]) {
#pragma unroll
        for (int i = 0; i < 4; ++i) {
            int e4 = tid + i * 256;
            int r = e4 >> 3, c4 = e4 & 7;
            v4bf w;
#pragma unroll
            for (int j = 0; j < 4; ++j) w[j] = (bf16_t)ar[i][j];
            *(v4bf*)(&As[p][r * LDSS + c4 * 4]) = w;
        }
    };
    auto loadB = [&](int k0, v4f br[4]) {
        if (TRANSB) {
#pragma unroll
            for (int i = 0; i < NI; ++i) {           // TBN rows of 32 k
                int e4 = tid + i * 256;
                int nc = e4 >> 3, c4 = e4 & 7;
                br[i] = *(const v4f*)(Bb + (long)(n0 + nc) * ldB + k0 + c4 * 4);
            }
        } else {
            if (TBN == 64 && tid >= 128) return;     // 128 micro-tiles only
            int kg = tid / (TBN / 4), ng = tid % (TBN / 4);  // 4x4 micro-tile
#pragma unroll
            for (int j = 0; j < 4; ++j)
                br[j] = *(const v4f*)(Bb + (long)(k0 + kg * 4 + j) * ldB + n0 + ng * 4);
        }
    };
    auto storeB = [&](int p, const v4f br[4]) {
        if (TRANSB) {
#pragma unroll
            for (int i = 0; i < NI; ++i) {
                int e4 = tid + i * 256;
                int nc = e4 >> 3, c4 = e4 & 7;
                v4bf w;
#pragma unroll
                for (int j = 0; j < 4; ++j) w[j] = (bf16_t)br[i][j];
                *(v4bf*)(&Bs[p][nc * LDSS + c4 * 4]) = w;
            }
        } else {
            if (TBN == 64 && tid >= 128) return;
            int kg = tid / (TBN / 4), ng = tid % (TBN / 4);
#pragma unroll
            for (int j = 0; j < 4; ++j) {            // register transpose 4x4
                v4bf w;
#pragma unroll
                for (int q = 0; q < 4; ++q) w[q] = (bf16_t)br[q][j];
                *(v4bf*)(&Bs[p][(ng * 4 + j) * LDSS + kg * 4]) = w;
            }
        }
    };

    v8f acc[2][NI];
#pragma unroll
    for (int mi = 0; mi < 2; ++mi)
#pragma unroll
        for (int ni = 0; ni < NI; ++ni) { v8f zv = {}; acc[mi][ni] = zv; }

    const int nk = Kd / BK;
    v4f arg[4], brg[4];

    // prologue: stage tile 0 into buffer 0
    loadA(0, arg);
    loadB(0, brg);
    storeA(0, arg);
    storeB(0, brg);

    for (int kb = 0; kb < nk; ++kb) {
        const int cur = kb & 1;
        const bool more = (kb + 1) < nk;
        if (more) {                      // issue next tile's global loads early
            loadA((kb + 1) * BK, arg);
            loadB((kb + 1) * BK, brg);
        }
        __syncthreads();                 // buffer 'cur' ready for all waves

        // ---- fragments per ISA 16-bit layouts ----
        v16bf afrag[2], bfrag[NI];
        const int kba = hb ? 8 : 0;      // A: lane-half -> K {0..7,16..23}/{8..15,24..31}
#pragma unroll
        for (int mi = 0; mi < 2; ++mi) {
            const bf16_t* ap = &As[cur][(wm * 32 + mi * 16 + lh) * LDSS];
            v8bf lo = *(const v8bf*)(ap + kba);
            v8bf hi = *(const v8bf*)(ap + 16 + kba);
            afrag[mi] = __builtin_shufflevector(lo, hi, 0,1,2,3,4,5,6,7,8,9,10,11,12,13,14,15);
        }
        const int kbb = hb * 8;          // B: lane-half -> K {0..7,16..23}/{8..15,24..31}
#pragma unroll
        for (int ni = 0; ni < NI; ++ni) {
            const bf16_t* bp = &Bs[cur][(wn * (TBN / 2) + ni * 16 + lh) * LDSS];
            v8bf lo = *(const v8bf*)(bp + kbb);
            v8bf hi = *(const v8bf*)(bp + 16 + kbb);
            bfrag[ni] = __builtin_shufflevector(lo, hi, 0,1,2,3,4,5,6,7,8,9,10,11,12,13,14,15);
        }
#pragma unroll
        for (int mi = 0; mi < 2; ++mi)
#pragma unroll
            for (int ni = 0; ni < NI; ++ni)
                acc[mi][ni] = __builtin_amdgcn_wmma_f32_16x16x32_bf16(
                    false, afrag[mi], false, bfrag[ni], (short)0, acc[mi][ni],
                    false, false);

        if (more) {                      // fill the other buffer (no conflict:
            storeA(cur ^ 1, arg);        //  readers of it synced at next barrier)
            storeB(cur ^ 1, brg);
        }
    }

    // ---- epilogue: C/D layout VGPR r -> row r (+8 for lane-half), col = lane%16
    const bool relu = (flags & FLAG_RELU) != 0;
#pragma unroll
    for (int mi = 0; mi < 2; ++mi)
#pragma unroll
        for (int ni = 0; ni < NI; ++ni)
#pragma unroll
            for (int r = 0; r < 8; ++r) {
                int row = m0 + wm * 32 + mi * 16 + hb * 8 + r;
                int col = n0 + wn * (TBN / 2) + ni * 16 + lh;
                float v = acc[mi][ni][r] * scale;
                if (bias) v += bias[col];
                if (Res)  v += Res[(long)row * ldC + col];
                if (relu) v = v > 0.0f ? v : 0.0f;
                Cb[(long)row * ldC + col] = v;
            }
}

// ---------------------------------------------------------------------------
// u[d] = sum_a sa_W[d,a] * sm_W[a]   (constant sa_b/sm_b term cancels in softmax)
// ---------------------------------------------------------------------------
__global__ void proj_u(const float* __restrict__ saW, const float* __restrict__ smW,
                       float* __restrict__ u)
{
    int d = blockIdx.x * 256 + threadIdx.x;
    if (d < 768) {
        float acc = 0.0f;
        for (int a = 0; a < 128; ++a) acc += saW[d * 128 + a] * smW[a];
        u[d] = acc;
    }
}

// ---------------------------------------------------------------------------
// Embedding gather + K=6 soft pooling + positional embedding.  1 WG per (b,s).
// ---------------------------------------------------------------------------
__global__ __launch_bounds__(256)
void embed_pool(const int* __restrict__ ids, const float* __restrict__ tok,
                const float* __restrict__ pos, const float* __restrict__ u,
                float* __restrict__ x)
{
    const int row = blockIdx.x;       // b*256 + s
    const int s   = row & 255;
    const int t   = threadIdx.x;
    __shared__ float red[6 * 256];
    __shared__ float alpha[6];

    float e[6][3];
#pragma unroll
    for (int k = 0; k < 6; ++k) {
        const int id = ids[row * 6 + k];
        const float* te = tok + (long)id * 768;
        float p = 0.0f;
#pragma unroll
        for (int j = 0; j < 3; ++j) {
            int d = t + j * 256;
            float v = te[d];
            e[k][j] = v;
            p += v * u[d];
        }
        red[k * 256 + t] = p;
    }
    __syncthreads();
    for (int st = 128; st > 0; st >>= 1) {
        if (t < st)
#pragma unroll
            for (int k = 0; k < 6; ++k) red[k * 256 + t] += red[k * 256 + t + st];
        __syncthreads();
    }
    if (t == 0) {
        float mx = -1e30f;
#pragma unroll
        for (int k = 0; k < 6; ++k) mx = fmaxf(mx, red[k * 256]);
        float ss = 0.0f;
#pragma unroll
        for (int k = 0; k < 6; ++k) { float a = expf(red[k * 256] - mx); alpha[k] = a; ss += a; }
        float inv = 1.0f / ss;
#pragma unroll
        for (int k = 0; k < 6; ++k) alpha[k] *= inv;
    }
    __syncthreads();
#pragma unroll
    for (int j = 0; j < 3; ++j) {
        int d = t + j * 256;
        float acc = pos[s * 768 + d];
#pragma unroll
        for (int k = 0; k < 6; ++k) acc += alpha[k] * e[k][j];
        x[(long)row * 768 + d] = acc;
    }
}

// ---------------------------------------------------------------------------
// Masked softmax over S=256 keys. One wave32 per row, 8 rows per block.
// ---------------------------------------------------------------------------
__global__ __launch_bounds__(256)
void attn_softmax(float* __restrict__ sc, const int* __restrict__ origin)
{
    const int row  = blockIdx.x * 8 + (threadIdx.x >> 5);  // < B*H*S = 98304
    const int lane = threadIdx.x & 31;
    const int b    = row / 3072;                           // H*S = 3072
    const long base = (long)row * 256;

    float v[8];
    float mx = -1e30f;
#pragma unroll
    for (int j = 0; j < 8; ++j) {
        int c = lane + j * 32;
        float val = sc[base + c];
        if (origin[b * 256 + c] == 0) val = -1.0e9f;
        v[j] = val;
        mx = fmaxf(mx, val);
    }
#pragma unroll
    for (int m = 16; m > 0; m >>= 1) mx = fmaxf(mx, __shfl_xor(mx, m, 32));
    float s = 0.0f;
#pragma unroll
    for (int j = 0; j < 8; ++j) { v[j] = __expf(v[j] - mx); s += v[j]; }
#pragma unroll
    for (int m = 16; m > 0; m >>= 1) s += __shfl_xor(s, m, 32);
    const float inv = 1.0f / s;
#pragma unroll
    for (int j = 0; j < 8; ++j) sc[base + lane + j * 32] = v[j] * inv;
}

// ---------------------------------------------------------------------------
// LayerNorm over D=768. One WG per row.
// ---------------------------------------------------------------------------
__global__ __launch_bounds__(256)
void layernorm(const float* __restrict__ in, const float* __restrict__ g,
               const float* __restrict__ bta, float* __restrict__ out)
{
    const long row = blockIdx.x;
    const int  t   = threadIdx.x;
    __shared__ float red[256];

    float v0 = in[row * 768 + t];
    float v1 = in[row * 768 + t + 256];
    float v2 = in[row * 768 + t + 512];
    red[t] = v0 + v1 + v2;
    __syncthreads();
    for (int st = 128; st > 0; st >>= 1) { if (t < st) red[t] += red[t + st]; __syncthreads(); }
    const float mu = red[0] * (1.0f / 768.0f);
    __syncthreads();
    const float d0 = v0 - mu, d1 = v1 - mu, d2 = v2 - mu;
    red[t] = d0 * d0 + d1 * d1 + d2 * d2;
    __syncthreads();
    for (int st = 128; st > 0; st >>= 1) { if (t < st) red[t] += red[t + st]; __syncthreads(); }
    const float inv = rsqrtf(red[0] * (1.0f / 768.0f) + 1e-5f);
    out[row * 768 + t]       = d0 * inv * g[t]       + bta[t];
    out[row * 768 + t + 256] = d1 * inv * g[t + 256] + bta[t + 256];
    out[row * 768 + t + 512] = d2 * inv * g[t + 512] + bta[t + 512];
}

// ---------------------------------------------------------------------------
// Head: rep = x[:,0,:]; logits = relu(rep@t1W+t1b)@t2W+t2b.
// d_out = [logits (32*2)] ++ [rep (32*768)]
// ---------------------------------------------------------------------------
__global__ __launch_bounds__(384)
void head(const float* __restrict__ x, const float* __restrict__ t1W,
          const float* __restrict__ t1b, const float* __restrict__ t2W,
          const float* __restrict__ t2b, float* __restrict__ out)
{
    const int b = blockIdx.x;
    const int t = threadIdx.x;                  // 0..383
    const float* rep = x + (long)b * 256 * 768; // row (b, s=0)
    out[64 + b * 768 + t]       = rep[t];
    out[64 + b * 768 + t + 384] = rep[t + 384];

    __shared__ float h1[384];
    float acc = t1b[t];
    for (int d = 0; d < 768; ++d) acc += rep[d] * t1W[d * 384 + t];
    h1[t] = fmaxf(acc, 0.0f);
    __syncthreads();
    if (t < 2) {
        float lg = t2b[t];
        for (int j = 0; j < 384; ++j) lg += h1[j] * t2W[j * 2 + t];
        out[b * 2 + t] = lg;
    }
}

// ---------------------------------------------------------------------------
extern "C" void kernel_launch(void* const* d_in, const int* in_sizes, int n_in,
                              void* d_out, int out_size, void* d_ws, size_t ws_size,
                              hipStream_t stream)
{
    (void)in_sizes; (void)n_in; (void)out_size; (void)ws_size;
    const int*   input_ids = (const int*)d_in[0];
    const int*   origin    = (const int*)d_in[1];
    const float* tok       = (const float*)d_in[2];
    const float* pos       = (const float*)d_in[3];
    const float* saW       = (const float*)d_in[4];
    // d_in[5] sa_b, d_in[7] sm_b: uniform shift before K-softmax -> cancels exactly
    const float* smW       = (const float*)d_in[6];
    const float* Wq = (const float*)d_in[8];  const float* bq = (const float*)d_in[9];
    const float* Wk = (const float*)d_in[10]; const float* bk = (const float*)d_in[11];
    const float* Wv = (const float*)d_in[12]; const float* bv = (const float*)d_in[13];
    const float* Wo = (const float*)d_in[14]; const float* bo = (const float*)d_in[15];
    const float* lng = (const float*)d_in[16]; const float* lnb = (const float*)d_in[17];
    const float* f1W = (const float*)d_in[18]; const float* f1b = (const float*)d_in[19];
    const float* f2W = (const float*)d_in[20]; const float* f2b = (const float*)d_in[21];
    const float* t1W = (const float*)d_in[22]; const float* t1b = (const float*)d_in[23];
    const float* t2W = (const float*)d_in[24]; const float* t2b = (const float*)d_in[25];

    const long XSZ = 8192L * 768;       // 6,291,456 floats
    float* ws  = (float*)d_ws;
    float* u   = ws;                    // 768 (padded to 4096)
    float* x   = ws + 4096;
    float* q   = x + XSZ;               // q, later ctx
    float* kb  = q + XSZ;               // k, later pre-LN tmp
    float* vb  = kb + XSZ;
    float* big = vb + XSZ;              // scores (B*H*S*S) / FFN mid (8192*3072)

    proj_u<<<3, 256, 0, stream>>>(saW, smW, u);
    embed_pool<<<8192, 256, 0, stream>>>(input_ids, tok, pos, u, x);

    for (int l = 0; l < 12; ++l) {
        const float* Wq_l = Wq + (long)l * 768 * 768;  const float* bq_l = bq + l * 768;
        const float* Wk_l = Wk + (long)l * 768 * 768;  const float* bk_l = bk + l * 768;
        const float* Wv_l = Wv + (long)l * 768 * 768;  const float* bv_l = bv + l * 768;
        const float* Wo_l = Wo + (long)l * 768 * 768;  const float* bo_l = bo + l * 768;
        const float* f1W_l = f1W + (long)l * 768 * 3072; const float* f1b_l = f1b + l * 3072;
        const float* f2W_l = f2W + (long)l * 3072 * 768; const float* f2b_l = f2b + l * 768;

        // Q, K, V projections: [8192,768] @ [768,768]
        gemm_wmma_bf16<128, false><<<dim3(6, 64, 1), 256, 0, stream>>>(
            x, Wq_l, q, bq_l, nullptr, 8192, 768, 768, 768, 768, 768,
            1, 0, 0, 0, 0, 0, 0, 1.0f, 0);
        gemm_wmma_bf16<128, false><<<dim3(6, 64, 1), 256, 0, stream>>>(
            x, Wk_l, kb, bk_l, nullptr, 8192, 768, 768, 768, 768, 768,
            1, 0, 0, 0, 0, 0, 0, 1.0f, 0);
        gemm_wmma_bf16<128, false><<<dim3(6, 64, 1), 256, 0, stream>>>(
            x, Wv_l, vb, bv_l, nullptr, 8192, 768, 768, 768, 768, 768,
            1, 0, 0, 0, 0, 0, 0, 1.0f, 0);

        // scores[b,h] = Q[b,:,h,:] @ K[b,:,h,:]^T * 0.125  (transB path)
        gemm_wmma_bf16<128, true><<<dim3(2, 2, 384), 256, 0, stream>>>(
            q, kb, big, nullptr, nullptr, 256, 256, 64, 768, 768, 256,
            12, 196608, 64, 196608, 64, 786432, 65536, 0.125f, 0);

        attn_softmax<<<12288, 256, 0, stream>>>(big, origin);

        // ctx[b,h] = attn @ V[b,:,h,:]  -> q buffer ([8192,768], col h*64), N=64 tile
        gemm_wmma_bf16<64, false><<<dim3(1, 2, 384), 256, 0, stream>>>(
            big, vb, q, nullptr, nullptr, 256, 64, 256, 256, 768, 768,
            12, 786432, 65536, 196608, 64, 196608, 64, 1.0f, 0);

        // out projection + bias + residual(x) -> kb; then LN -> x
        gemm_wmma_bf16<128, false><<<dim3(6, 64, 1), 256, 0, stream>>>(
            q, Wo_l, kb, bo_l, x, 8192, 768, 768, 768, 768, 768,
            1, 0, 0, 0, 0, 0, 0, 1.0f, 0);
        layernorm<<<8192, 256, 0, stream>>>(kb, lng + l * 768, lnb + l * 768, x);

        // FFN: relu(x@f1W+f1b) -> big; big@f2W+f2b -> x
        gemm_wmma_bf16<128, false><<<dim3(24, 64, 1), 256, 0, stream>>>(
            x, f1W_l, big, f1b_l, nullptr, 8192, 3072, 768, 768, 3072, 3072,
            1, 0, 0, 0, 0, 0, 0, 1.0f, FLAG_RELU);
        gemm_wmma_bf16<128, false><<<dim3(6, 64, 1), 256, 0, stream>>>(
            big, f2W_l, x, f2b_l, nullptr, 8192, 768, 3072, 3072, 768, 768,
            1, 0, 0, 0, 0, 0, 0, 1.0f, 0);
    }

    head<<<32, 384, 0, stream>>>(x, t1W, t1b, t2W, t2b, (float*)d_out);
}
